// SuperGATConvEdgeClassifier_86938728005824
// MI455X (gfx1250) — compile-verified, matrix operand order checked by reading, not run on hardware
//
#include <hip/hip_runtime.h>

#define HID 128

typedef __attribute__((ext_vector_type(16))) _Float16 v16h;
typedef __attribute__((ext_vector_type(8)))  _Float16 v8h;
typedef __attribute__((ext_vector_type(8)))  float    v8f;

// ---- ordered-uint encoding for float atomic max ----
__device__ __forceinline__ unsigned f2ord(float f) {
    unsigned u = __float_as_uint(f);
    return (u & 0x80000000u) ? ~u : (u | 0x80000000u);
}
__device__ __forceinline__ float ord2f(unsigned v) {
    return __uint_as_float((v & 0x80000000u) ? (v & 0x7fffffffu) : ~v);
}

// Pack W [128,128] row-major f32 into per-lane WMMA B fragments (f16).
// Fragment frag = nt*4+kt; lane holds column n = nt*16 + lane%16,
// K values kt*32 + base + {0..7} and kt*32 + base + 16 + {0..7}, base = lane<16?0:8.
__global__ void k_pack_w(const float* __restrict__ W, _Float16* __restrict__ bp) {
    int frag = blockIdx.x;          // 0..31
    int lane = threadIdx.x;         // 0..31
    int nt = frag >> 2, kt = frag & 3;
    int n = nt * 16 + (lane & 15);
    int kbase = kt * 32 + ((lane < 16) ? 0 : 8);
    _Float16* out = bp + ((size_t)frag * 32 + lane) * 16;
    for (int j = 0; j < 8; ++j) {
        out[j]     = (_Float16)W[(kbase + j) * HID + n];
        out[8 + j] = (_Float16)W[(kbase + 16 + j) * HID + n];
    }
}

__global__ void k_x2h(const float* __restrict__ x, _Float16* __restrict__ xh, int total) {
    int i = blockIdx.x * blockDim.x + threadIdx.x;
    if (i < total) xh[i] = (_Float16)x[i];
}

__global__ void k_init(float* __restrict__ agg, unsigned* __restrict__ amax,
                       float* __restrict__ denom, int N) {
    int i = blockIdx.x * blockDim.x + threadIdx.x;
    if (i < N * HID) agg[i] = 0.0f;
    if (i < N) { amax[i] = 0u; denom[i] = 0.0f; }
}

// C[M,128] = A[M,128] @ B[128,128]  via v_wmma_f32_16x16x32_f16, f32 accumulate.
// B fragments are DMA'd into LDS with CDNA5 async load-to-LDS; C tiles are
// staged through LDS for coalesced b128 stores.
__global__ __launch_bounds__(128) void k_gemm(const _Float16* __restrict__ A,
                                              const _Float16* __restrict__ Bp,
                                              float* __restrict__ C,
                                              int M, int Mtiles) {
    __shared__ v16h  ldsB[32][32];      // [frag = nt*4+kt][lane], 32 KB
    __shared__ float ldsC[4][16][16];   // per-wave C staging, 4 KB
    int tid = threadIdx.x;

    // ---- async DMA of the 32 KB packed-B into LDS (16 B per lane per issue) ----
    {
        const char* gB = (const char*)Bp;
        char* lBgen = (char*)&ldsB[0][0];
        for (int i = tid; i < 2048; i += 128) {
            unsigned long long gaddr = (unsigned long long)(gB + (size_t)i * 16);
            unsigned loff = (unsigned)(unsigned long long)
                (__attribute__((address_space(3))) char*)(lBgen + (size_t)i * 16);
            asm volatile("global_load_async_to_lds_b128 %0, %1, off"
                         :: "v"(loff), "v"(gaddr) : "memory");
        }
        asm volatile("s_wait_asynccnt 0" ::: "memory");
    }
    __syncthreads();

    int wid = tid >> 5, lane = tid & 31;
    int rt = blockIdx.x * 4 + wid;       // wave-uniform tile id (EXEC stays full)
    if (rt >= Mtiles) return;

    int row = rt * 16 + (lane & 15);
    if (row >= M) row = M - 1;
    int base = (lane < 16) ? 0 : 8;

    v16h a[4];
    const _Float16* arp = A + (size_t)row * HID + base;
    for (int kt = 0; kt < 4; ++kt) {
        const _Float16* ap = arp + kt * 32;
        v8h lo = *(const v8h*)(ap);
        v8h hi = *(const v8h*)(ap + 16);
        for (int j = 0; j < 8; ++j) { a[kt][j] = lo[j]; a[kt][8 + j] = hi[j]; }
    }

    int mrow = (lane < 16) ? 0 : 8;
    int colL = lane & 15;
    int r0 = lane >> 1;                    // output row handled by this lane
    int cg = (lane & 1) * 8;               // output column group (0 or 8)
    int rr = rt * 16 + r0;

    if ((rt * 16 + 16) <= M) {
        // ---- hot path: full tile, unconditional coalesced stores ----
        for (int nt = 0; nt < 8; ++nt) {
            v8f c = {};
            for (int kt = 0; kt < 4; ++kt) {
                c = __builtin_amdgcn_wmma_f32_16x16x32_f16(
                        false, a[kt], false, ldsB[nt * 4 + kt][lane],
                        (short)0, c, false, false);
            }
            #pragma unroll
            for (int r = 0; r < 8; ++r) ldsC[wid][mrow + r][colL] = c[r];
            asm volatile("s_wait_dscnt 0" ::: "memory");
            float4 lo4 = *(float4*)&ldsC[wid][r0][cg];
            float4 hi4 = *(float4*)&ldsC[wid][r0][cg + 4];
            asm volatile("" ::: "memory");   // keep reads before next iter's stores
            float* cp = C + (size_t)rr * HID + nt * 16 + cg;
            *(float4*)cp = lo4;
            *((float4*)cp + 1) = hi4;
        }
    } else {
        // ---- tail path: per-row guard ----
        for (int nt = 0; nt < 8; ++nt) {
            v8f c = {};
            for (int kt = 0; kt < 4; ++kt) {
                c = __builtin_amdgcn_wmma_f32_16x16x32_f16(
                        false, a[kt], false, ldsB[nt * 4 + kt][lane],
                        (short)0, c, false, false);
            }
            #pragma unroll
            for (int r = 0; r < 8; ++r) ldsC[wid][mrow + r][colL] = c[r];
            asm volatile("s_wait_dscnt 0" ::: "memory");
            float4 lo4 = *(float4*)&ldsC[wid][r0][cg];
            float4 hi4 = *(float4*)&ldsC[wid][r0][cg + 4];
            asm volatile("" ::: "memory");
            if (rr < M) {
                float* cp = C + (size_t)rr * HID + nt * 16 + cg;
                *(float4*)cp = lo4;
                *((float4*)cp + 1) = hi4;
            }
        }
    }
}

// Wave per edge: alpha = (hj.att_l + hi.att_r) * sigmoid(hi.hj); atomic segment-max.
__global__ __launch_bounds__(256) void k_alpha(const float* __restrict__ h,
                                               const int* __restrict__ ei,
                                               const float* __restrict__ attl,
                                               const float* __restrict__ attr,
                                               float* __restrict__ alpha,
                                               unsigned* __restrict__ amax,
                                               int E, int Etot) {
    int e = blockIdx.x * 8 + (threadIdx.x >> 5);
    int lane = threadIdx.x & 31;
    if (e >= Etot) return;
    int s, d;
    if (e < E) { s = ei[e]; d = ei[E + e]; } else { s = e - E; d = s; }
    float4 hj = ((const float4*)(h + (size_t)s * HID))[lane];
    float4 hi = ((const float4*)(h + (size_t)d * HID))[lane];
    float4 al = ((const float4*)attl)[lane];
    float4 arr = ((const float4*)attr)[lane];
    float logit = hj.x*hi.x + hj.y*hi.y + hj.z*hi.z + hj.w*hi.w;
    float lin = hj.x*al.x + hj.y*al.y + hj.z*al.z + hj.w*al.w
              + hi.x*arr.x + hi.y*arr.y + hi.z*arr.z + hi.w*arr.w;
    for (int off = 16; off; off >>= 1) {
        logit += __shfl_xor(logit, off, 32);
        lin   += __shfl_xor(lin, off, 32);
    }
    if (lane == 0) {
        float sig = 1.0f / (1.0f + __expf(-logit));
        float av = lin * sig;
        alpha[e] = av;
        atomicMax(&amax[d], f2ord(av));
    }
}

__global__ void k_exp(float* __restrict__ alpha, const int* __restrict__ ei,
                      const unsigned* __restrict__ amax, float* __restrict__ denom,
                      int E, int Etot) {
    int e = blockIdx.x * blockDim.x + threadIdx.x;
    if (e >= Etot) return;
    int d = (e < E) ? ei[E + e] : (e - E);
    float ea = __expf(alpha[e] - ord2f(amax[d]));
    alpha[e] = ea;
    atomicAdd(&denom[d], ea);
}

// Wave per edge: agg[dst] += hj * (ea / (denom+eps)) with f32 atomics.
__global__ __launch_bounds__(256) void k_scatter(const float* __restrict__ h,
                                                 const int* __restrict__ ei,
                                                 const float* __restrict__ ea,
                                                 const float* __restrict__ denom,
                                                 float* __restrict__ agg,
                                                 int E, int Etot) {
    int e = blockIdx.x * 8 + (threadIdx.x >> 5);
    int lane = threadIdx.x & 31;
    if (e >= Etot) return;
    int s, d;
    if (e < E) { s = ei[e]; d = ei[E + e]; } else { s = e - E; d = s; }
    float w = ea[e] / (denom[d] + 1e-16f);
    float4 hj = ((const float4*)(h + (size_t)s * HID))[lane];
    float* p = agg + (size_t)d * HID + lane * 4;
    atomicAdd(p + 0, hj.x * w);
    atomicAdd(p + 1, hj.y * w);
    atomicAdd(p + 2, hj.z * w);
    atomicAdd(p + 3, hj.w * w);
}

// Layer-1 epilogue: +bias, ReLU, write f16 for next GEMM, re-zero buffers.
__global__ void k_epi1(float* __restrict__ agg, const float* __restrict__ b,
                       _Float16* __restrict__ o16,
                       unsigned* __restrict__ amax, float* __restrict__ denom, int N) {
    int i = blockIdx.x * blockDim.x + threadIdx.x;
    if (i >= N * HID) return;
    float v = agg[i] + b[i & (HID - 1)];
    v = fmaxf(v, 0.0f);
    o16[i] = (_Float16)v;
    agg[i] = 0.0f;
    if ((i & (HID - 1)) == 0) { int n = i >> 7; amax[n] = 0u; denom[n] = 0.0f; }
}

// Layer-2 epilogue: +bias, f32 output for classifier gathers.
__global__ void k_epi2(const float* __restrict__ agg, const float* __restrict__ b,
                       float* __restrict__ h2, int N) {
    int i = blockIdx.x * blockDim.x + threadIdx.x;
    if (i >= N * HID) return;
    h2[i] = agg[i] + b[i & (HID - 1)];
}

// Wave per edge: out = [h[row], h[col]] @ Wc + bc  (Wc is [256,4] row-major).
__global__ __launch_bounds__(256) void k_cls(const float* __restrict__ h2,
                                             const int* __restrict__ ei,
                                             const float* __restrict__ Wc,
                                             const float* __restrict__ bc,
                                             float* __restrict__ out, int E) {
    int e = blockIdx.x * 8 + (threadIdx.x >> 5);
    int lane = threadIdx.x & 31;
    if (e >= E) return;
    int s = ei[e], d = ei[E + e];
    const float4* wr = (const float4*)Wc;
    float4 hs = ((const float4*)(h2 + (size_t)s * HID))[lane];
    float4 hd = ((const float4*)(h2 + (size_t)d * HID))[lane];
    int c0 = lane * 4;
    float hv[4] = {hs.x, hs.y, hs.z, hs.w};
    float gv[4] = {hd.x, hd.y, hd.z, hd.w};
    float4 acc = make_float4(0.f, 0.f, 0.f, 0.f);
    for (int j = 0; j < 4; ++j) {
        float4 w1 = wr[c0 + j];
        float4 w2 = wr[HID + c0 + j];
        acc.x += hv[j] * w1.x + gv[j] * w2.x;
        acc.y += hv[j] * w1.y + gv[j] * w2.y;
        acc.z += hv[j] * w1.z + gv[j] * w2.z;
        acc.w += hv[j] * w1.w + gv[j] * w2.w;
    }
    for (int off = 16; off; off >>= 1) {
        acc.x += __shfl_xor(acc.x, off, 32);
        acc.y += __shfl_xor(acc.y, off, 32);
        acc.z += __shfl_xor(acc.z, off, 32);
        acc.w += __shfl_xor(acc.w, off, 32);
    }
    if (lane == 0) {
        float4 r = make_float4(acc.x + bc[0], acc.y + bc[1], acc.z + bc[2], acc.w + bc[3]);
        ((float4*)out)[e] = r;
    }
}

extern "C" void kernel_launch(void* const* d_in, const int* in_sizes, int n_in,
                              void* d_out, int out_size, void* d_ws, size_t ws_size,
                              hipStream_t stream) {
    const float* x   = (const float*)d_in[0];
    const int*   ei  = (const int*)d_in[1];
    const float* W1  = (const float*)d_in[2];
    const float* al1 = (const float*)d_in[3];
    const float* ar1 = (const float*)d_in[4];
    const float* b1  = (const float*)d_in[5];
    const float* W2  = (const float*)d_in[6];
    const float* al2 = (const float*)d_in[7];
    const float* ar2 = (const float*)d_in[8];
    const float* b2  = (const float*)d_in[9];
    const float* Wc  = (const float*)d_in[10];
    const float* bc  = (const float*)d_in[11];
    (void)n_in; (void)out_size; (void)ws_size;

    int N = in_sizes[0] / HID;
    int E = in_sizes[1] / 2;
    int Etot = E + N;
    int Mtiles = (N + 15) / 16;
    int tNH = N * HID;

    char* ws = (char*)d_ws;
    size_t off = 0;
    auto take = [&](size_t bytes) {
        char* p = ws + off;
        off = (off + bytes + 255) & ~(size_t)255;
        return p;
    };
    _Float16* xh    = (_Float16*)take((size_t)tNH * 2);
    _Float16* bp1   = (_Float16*)take(32 * 32 * 16 * 2);
    _Float16* bp2   = (_Float16*)take(32 * 32 * 16 * 2);
    float*    hlin  = (float*)take((size_t)tNH * 4);
    float*    agg   = (float*)take((size_t)tNH * 4);
    float*    alpha = (float*)take((size_t)Etot * 4);
    unsigned* amax  = (unsigned*)take((size_t)N * 4);
    float*    den   = (float*)take((size_t)N * 4);

    dim3 b256(256);
    int gElem = (tNH + 255) / 256;
    int gEdgeT = (Etot + 255) / 256;
    int gWave = (Etot + 7) / 8;
    int gGemm = (Mtiles + 3) / 4;

    k_pack_w<<<32, 32, 0, stream>>>(W1, bp1);
    k_pack_w<<<32, 32, 0, stream>>>(W2, bp2);
    k_x2h<<<gElem, b256, 0, stream>>>(x, xh, tNH);
    k_init<<<gElem, b256, 0, stream>>>(agg, amax, den, N);

    // ---- Layer 1 ----
    k_gemm<<<gGemm, 128, 0, stream>>>(xh, bp1, hlin, N, Mtiles);
    k_alpha<<<gWave, b256, 0, stream>>>(hlin, ei, al1, ar1, alpha, amax, E, Etot);
    k_exp<<<gEdgeT, b256, 0, stream>>>(alpha, ei, amax, den, E, Etot);
    k_scatter<<<gWave, b256, 0, stream>>>(hlin, ei, alpha, den, agg, E, Etot);
    k_epi1<<<gElem, b256, 0, stream>>>(agg, b1, xh, amax, den, N);

    // ---- Layer 2 ----
    k_gemm<<<gGemm, 128, 0, stream>>>(xh, bp2, hlin, N, Mtiles);
    k_alpha<<<gWave, b256, 0, stream>>>(hlin, ei, al2, ar2, alpha, amax, E, Etot);
    k_exp<<<gEdgeT, b256, 0, stream>>>(alpha, ei, amax, den, E, Etot);
    k_scatter<<<gWave, b256, 0, stream>>>(hlin, ei, alpha, den, agg, E, Etot);
    k_epi2<<<gElem, b256, 0, stream>>>(agg, b2, hlin, N);

    // ---- Edge classifier ----
    k_cls<<<(E + 7) / 8, b256, 0, stream>>>(hlin, ei, Wc, bc, (float*)d_out, E);
}
